// PointTransformer_42099269435613
// MI455X (gfx1250) — compile-verified, matrix operand order checked by reading, not run on hardware
//
#include <hip/hip_runtime.h>
#include <math.h>

// ---------------------------------------------------------------------------
// Types for CDNA5 WMMA (wave32)
// ---------------------------------------------------------------------------
typedef __bf16 bf16;
typedef __attribute__((ext_vector_type(16))) __bf16        v16bf;
typedef __attribute__((ext_vector_type(8)))  float         v8f;
typedef __attribute__((ext_vector_type(8)))  unsigned int  v8u;

union FragCast { v8u u; v16bf b; };
union Pack4 { bf16 h[4]; unsigned long long u; };

#define B_BATCH 4
#define NPTS    4096
#define CDIM    256
#define BNROWS  (B_BATCH * NPTS)   // 16384
#define KNN     16
#define HEADD   64

// ---------------------------------------------------------------------------
// CDNA5 async global->LDS copy (ASYNCcnt) helpers
// ---------------------------------------------------------------------------
__device__ __forceinline__ void async_b128(const bf16* gaddr, unsigned lds_off) {
    asm volatile("global_load_async_to_lds_b128 %0, %1, off"
                 :: "v"(lds_off), "v"(gaddr) : "memory");
}
__device__ __forceinline__ void wait_async0() {
#if __has_builtin(__builtin_amdgcn_s_wait_asynccnt)
    __builtin_amdgcn_s_wait_asynccnt(0);
#else
    asm volatile("s_wait_asynccnt 0x0" ::: "memory");
#endif
}

// ---------------------------------------------------------------------------
// Wave reduction helper (wave32)
// ---------------------------------------------------------------------------
__device__ __forceinline__ float wave_sum(float v) {
#pragma unroll
    for (int off = 16; off > 0; off >>= 1) v += __shfl_xor(v, off, 32);
    return v;
}

// ---------------------------------------------------------------------------
// WMMA GEMM, bf16 inputs:  C[M,Nd] = act( A[M,Kd] @ Bw[Nd,Kd]^T + bias )
// Block tile 128x128, K-step 32, 8 waves in 2(M)x4(N), each wave 64x32
// (4x2 accumulators -> 8 WMMA per K-step from 4 A-frags + 2 B-frags).
// Double-buffered LDS tiles filled by async global->LDS b128 copies:
// the next K-tile streams in while WMMAs consume the current one.
// Output: fp32 (C32) or bf16 (C16), whichever pointer is non-null.
// M%128==0, Nd%128==0, Kd%32==0 guaranteed by caller.
// ---------------------------------------------------------------------------
__global__ __launch_bounds__(256)
void gemm_bf16_wmma(float* __restrict__ C32, bf16* __restrict__ C16,
                    const bf16* __restrict__ A, const bf16* __restrict__ Bw,
                    const float* __restrict__ bias,
                    int M, int Nd, int Kd, int relu)
{
    __shared__ __align__(16) bf16 As[2][128][32];   // 2 x 8 KB
    __shared__ __align__(16) bf16 Bs[2][128][32];   // 2 x 8 KB

    const int tid  = threadIdx.x;
    const int lane = tid & 31;
    const int wid  = tid >> 5;
    const int wm   = wid & 1;           // 0..1  -> 64-row slab
    const int wn   = wid >> 1;          // 0..3  -> 32-col slab
    const int m0   = blockIdx.y * 128;
    const int n0   = blockIdx.x * 128;
    const int half = lane >> 4;         // 0/1
    const int r16  = lane & 15;

    // --- per-thread staging addresses (hoisted out of the K loop) ---
    // each tile = 512 16B chunks; thread owns chunks {tid, tid+256}
    // -> same column group, rows r and r+64
    const int  c8  = (tid & 3) * 8;                 // element column
    const int  r   = tid >> 2;                      // 0..63
    const bf16* gA0 = A  + (long)(m0 + r)      * Kd + c8;
    const bf16* gA1 = A  + (long)(m0 + r + 64) * Kd + c8;
    const bf16* gB0 = Bw + (long)(n0 + r)      * Kd + c8;
    const bf16* gB1 = Bw + (long)(n0 + r + 64) * Kd + c8;
    const unsigned lA0 = (unsigned)(size_t)&As[0][0][0] + (unsigned)(r * 64 + (tid & 3) * 16);
    const unsigned lA1 = lA0 + 64 * 64;
    const unsigned lB0 = (unsigned)(size_t)&Bs[0][0][0] + (unsigned)(r * 64 + (tid & 3) * 16);
    const unsigned lB1 = lB0 + 64 * 64;

    auto stage = [&](int k0, int buf) {
        const unsigned bo = (unsigned)buf * 8192u;
        async_b128(gA0 + k0, lA0 + bo);
        async_b128(gA1 + k0, lA1 + bo);
        async_b128(gB0 + k0, lB0 + bo);
        async_b128(gB1 + k0, lB1 + bo);
    };

    v8f acc[4][2] = {};

    const int nk = Kd >> 5;
    stage(0, 0);
    for (int ki = 0; ki < nk; ++ki) {
        const int buf = ki & 1;
        wait_async0();
        __syncthreads();
        if (ki + 1 < nk) stage((ki + 1) << 5, 1 - buf);

        const bf16 (*Asb)[32] = As[buf];
        const bf16 (*Bsb)[32] = Bs[buf];

        v16bf af[4], bfrag[2];
#pragma unroll
        for (int mt = 0; mt < 4; ++mt) {
            const int row = wm * 64 + mt * 16 + r16;
            FragCast f;
#pragma unroll
            for (int v = 0; v < 8; ++v) {
                const int kk = ((v & 4) << 2) + half * 8 + ((v & 3) << 1);
                f.u[v] = *(const unsigned int*)&Asb[row][kk];
            }
            af[mt] = f.b;
        }
#pragma unroll
        for (int nt = 0; nt < 2; ++nt) {
            const int row = wn * 32 + nt * 16 + r16;
            FragCast f;
#pragma unroll
            for (int v = 0; v < 8; ++v) {
                const int kk = ((v & 4) << 2) + half * 8 + ((v & 3) << 1);
                f.u[v] = *(const unsigned int*)&Bsb[row][kk];
            }
            bfrag[nt] = f.b;
        }
#pragma unroll
        for (int mt = 0; mt < 4; ++mt)
#pragma unroll
            for (int nt = 0; nt < 2; ++nt)
                acc[mt][nt] = __builtin_amdgcn_wmma_f32_16x16x32_bf16(
                    false, af[mt], false, bfrag[nt], (short)0, acc[mt][nt],
                    false, false);
    }

    // epilogue: D layout — VGPR g, lanes 0-15: M=g; lanes 16-31: M=8+g
#pragma unroll
    for (int mt = 0; mt < 4; ++mt)
#pragma unroll
        for (int nt = 0; nt < 2; ++nt) {
            const int gm = m0 + wm * 64 + mt * 16 + half * 8;
            const int gn = n0 + wn * 32 + nt * 16 + r16;
            const float bb = bias ? bias[gn] : 0.f;
#pragma unroll
            for (int g = 0; g < 8; ++g) {
                float x = acc[mt][nt][g] + bb;
                if (relu) x = fmaxf(x, 0.f);
                if (C16) C16[(long)(gm + g) * Nd + gn] = (bf16)x;
                else     C32[(long)(gm + g) * Nd + gn] = x;
            }
        }
}

// ---------------------------------------------------------------------------
// fp32 -> bf16 elementwise convert (vec4). n must be divisible by 4.
// ---------------------------------------------------------------------------
__global__ __launch_bounds__(256)
void cvt_bf16_kernel(bf16* __restrict__ out, const float* __restrict__ in, long n4)
{
    const long t = (long)blockIdx.x * 256 + threadIdx.x;
    if (t >= n4) return;
    const float4 v = ((const float4*)in)[t];
    Pack4 pk;
    pk.h[0] = (bf16)v.x; pk.h[1] = (bf16)v.y;
    pk.h[2] = (bf16)v.z; pk.h[3] = (bf16)v.w;
    ((unsigned long long*)out)[t] = pk.u;
}

// ---------------------------------------------------------------------------
// LayerNorm over C=256 with optional residual add; optional bf16 mirror out.
// One wave per row.
// ---------------------------------------------------------------------------
__global__ __launch_bounds__(256)
void layernorm_kernel(float* __restrict__ out, bf16* __restrict__ out16,
                      const float* __restrict__ x, const float* __restrict__ res,
                      const float* __restrict__ g, const float* __restrict__ beta)
{
    const int row  = blockIdx.x * 8 + (threadIdx.x >> 5);
    const int lane = threadIdx.x & 31;
    const float* xr = x + (long)row * CDIM;
    const float* rr = res ? res + (long)row * CDIM : nullptr;

    float v[8];
    float s = 0.f;
#pragma unroll
    for (int i = 0; i < 8; ++i) {
        float t = xr[lane + 32 * i];
        if (rr) t += rr[lane + 32 * i];
        v[i] = t; s += t;
    }
    const float mean = wave_sum(s) * (1.0f / CDIM);
    float q = 0.f;
#pragma unroll
    for (int i = 0; i < 8; ++i) { float d = v[i] - mean; q += d * d; }
    const float rstd = rsqrtf(wave_sum(q) * (1.0f / CDIM) + 1e-5f);
#pragma unroll
    for (int i = 0; i < 8; ++i) {
        const int c = lane + 32 * i;
        const float y = (v[i] - mean) * rstd * g[c] + beta[c];
        out[(long)row * CDIM + c] = y;
        if (out16) out16[(long)row * CDIM + c] = (bf16)y;
    }
}

// sum of squares per row (for KNN distance). One wave per row.
__global__ __launch_bounds__(256)
void rowsumsq_kernel(float* __restrict__ sq, const float* __restrict__ h)
{
    const int row  = blockIdx.x * 8 + (threadIdx.x >> 5);
    const int lane = threadIdx.x & 31;
    const float* hr = h + (long)row * CDIM;
    float s = 0.f;
#pragma unroll
    for (int i = 0; i < 8; ++i) { float t = hr[lane + 32 * i]; s += t * t; }
    s = wave_sum(s);
    if (lane == 0) sq[row] = s;
}

// ---------------------------------------------------------------------------
// Top-16 smallest (sq[m] - 2*dot[n,m]) over m in [0,4096). One block per row.
// ---------------------------------------------------------------------------
__global__ __launch_bounds__(256)
void topk_kernel(int* __restrict__ idx_out, const float* __restrict__ dot,
                 const float* __restrict__ sq)
{
    __shared__ float key[NPTS];         // 16 KB
    __shared__ float sval[256];
    __shared__ int   sidx[256];

    const int n = blockIdx.x;
    const float* dr = dot + (long)n * NPTS;
    for (int m = threadIdx.x; m < NPTS; m += 256)
        key[m] = sq[m] - 2.0f * dr[m];
    __syncthreads();

    for (int t = 0; t < KNN; ++t) {
        float bv = 3.4e38f; int bi = 0;
        for (int m = threadIdx.x; m < NPTS; m += 256) {
            float k = key[m];
            if (k < bv) { bv = k; bi = m; }
        }
        sval[threadIdx.x] = bv; sidx[threadIdx.x] = bi;
        __syncthreads();
        for (int s = 128; s > 0; s >>= 1) {
            if (threadIdx.x < s) {
                if (sval[threadIdx.x + s] < sval[threadIdx.x]) {
                    sval[threadIdx.x] = sval[threadIdx.x + s];
                    sidx[threadIdx.x] = sidx[threadIdx.x + s];
                }
            }
            __syncthreads();
        }
        if (threadIdx.x == 0) { idx_out[t] = sidx[0]; key[sidx[0]] = 3.4e38f; }
        __syncthreads();
    }
}

// ---------------------------------------------------------------------------
// Gathered neighborhood attention: one wave per (point, head).
// ---------------------------------------------------------------------------
__global__ __launch_bounds__(128)
void attn_kernel(float* __restrict__ out, const float* __restrict__ Q,
                 const float* __restrict__ Km, const float* __restrict__ Vm,
                 const int* __restrict__ idx)
{
    const int bn   = blockIdx.x;          // 0..B*N-1
    const int h    = threadIdx.x >> 5;    // 0..3
    const int lane = threadIdx.x & 31;
    const int b    = bn >> 12;            // /4096
    const long base = (long)bn * CDIM + h * HEADD;

    const float q0 = Q[base + lane];
    const float q1 = Q[base + 32 + lane];

    float s[KNN], v0[KNN], v1[KNN];
#pragma unroll
    for (int j = 0; j < KNN; ++j) {
        const int m = idx[(long)bn * KNN + j];
        const long rb = ((long)(b << 12) + m) * CDIM + h * HEADD;
        const float k0 = Km[rb + lane];
        const float k1 = Km[rb + 32 + lane];
        v0[j] = Vm[rb + lane];
        v1[j] = Vm[rb + 32 + lane];
        s[j] = wave_sum(q0 * k0 + q1 * k1) * 0.125f;   // / sqrt(64)
    }
    float mx = s[0];
#pragma unroll
    for (int j = 1; j < KNN; ++j) mx = fmaxf(mx, s[j]);
    float sum = 0.f;
#pragma unroll
    for (int j = 0; j < KNN; ++j) { s[j] = __expf(s[j] - mx); sum += s[j]; }
    const float inv = 1.0f / sum;
    float o0 = 0.f, o1 = 0.f;
#pragma unroll
    for (int j = 0; j < KNN; ++j) { o0 += s[j] * v0[j]; o1 += s[j] * v1[j]; }
    out[base + lane]      = o0 * inv;
    out[base + 32 + lane] = o1 * inv;
}

// ---------------------------------------------------------------------------
// Max-pool over N per (batch, channel).
// ---------------------------------------------------------------------------
__global__ __launch_bounds__(256)
void maxpool_kernel(float* __restrict__ out, const float* __restrict__ H)
{
    const int t = blockIdx.x * blockDim.x + threadIdx.x;
    if (t >= B_BATCH * CDIM) return;
    const int b = t >> 8, c = t & 255;
    const float* p = H + ((long)b * NPTS) * CDIM + c;
    float m = -3.4e38f;
    for (int n = 0; n < NPTS; ++n) m = fmaxf(m, p[(long)n * CDIM]);
    out[t] = m;
}

// ---------------------------------------------------------------------------
// Small row-wise linear (+BN eval +ReLU) for embed / FC head. One thread/out.
// ---------------------------------------------------------------------------
__global__ __launch_bounds__(256)
void rowlinear_kernel(float* __restrict__ out, const float* __restrict__ in,
                      const float* __restrict__ W, const float* __restrict__ bias,
                      const float* __restrict__ bn_g, const float* __restrict__ bn_b,
                      int relu, int M, int Cin, int Cout)
{
    const int t = blockIdx.x * blockDim.x + threadIdx.x;
    if (t >= M * Cout) return;
    const int o = t % Cout;
    const int m = t / Cout;
    float acc = bias ? bias[o] : 0.f;
    const float* ir = in + (long)m * Cin;
    const float* wr = W + (long)o * Cin;
    for (int k = 0; k < Cin; ++k) acc += ir[k] * wr[k];
    if (bn_g) acc = acc * (bn_g[o] * rsqrtf(1.0f + 1e-5f)) + bn_b[o];
    if (relu) acc = fmaxf(acc, 0.f);
    out[t] = acc;
}

// ---------------------------------------------------------------------------
// Host-side orchestration
// ---------------------------------------------------------------------------
extern "C" void kernel_launch(void* const* d_in, const int* in_sizes, int n_in,
                              void* d_out, int out_size, void* d_ws, size_t ws_size,
                              hipStream_t stream)
{
    (void)in_sizes; (void)out_size; (void)ws_size;

    // --- input map (dict order from setup_inputs) ---
    const float* P[91];
    for (int i = 0; i < n_in && i < 91; ++i) P[i] = (const float*)d_in[i];
    const float* x = P[0];

    // --- workspace carve-out ---
    char* wp = (char*)d_ws;
    auto carve = [&](size_t bytes) -> void* {
        void* p = (void*)wp;
        wp += (bytes + 255) & ~(size_t)255;
        return p;
    };
    const size_t act = (size_t)BNROWS * CDIM * sizeof(float);   // 16 MB
    float* H    = (float*)carve(act);
    float* H2   = (float*)carve(act);
    float* T    = (float*)carve(act);
    float* Qb   = (float*)carve(act);
    float* Kb   = (float*)carve(act);
    float* Vb   = (float*)carve(act);
    float* AO   = (float*)carve(act);
    float* BIG  = (float*)carve((size_t)NPTS * NPTS * sizeof(float)); // 64 MB (dist)
    bf16*  Hbf  = (bf16*)carve((size_t)BNROWS * CDIM * 2);            // 8 MB
    bf16*  H2bf = (bf16*)carve((size_t)BNROWS * CDIM * 2);            // 8 MB
    bf16*  FHbf = (bf16*)carve((size_t)BNROWS * 1024 * 2);            // 32 MB (ffn hidden)
    bf16*  E2bf = (bf16*)carve((size_t)BNROWS * 128 * 2);             // 4 MB
    float* SQ   = (float*)carve((size_t)BNROWS * sizeof(float));
    int*   IDX  = (int*)  carve((size_t)BNROWS * KNN * sizeof(int));
    float* E1   = (float*)carve((size_t)BNROWS * 64  * sizeof(float));
    float* E2   = (float*)carve((size_t)BNROWS * 128 * sizeof(float));
    float* GP   = (float*)carve((size_t)B_BATCH * 256 * sizeof(float));
    float* F1   = (float*)carve((size_t)B_BATCH * 512 * sizeof(float));
    float* F2   = (float*)carve((size_t)B_BATCH * 256 * sizeof(float));
    // bf16 weight staging (reused each block)
    bf16* Wbf_in = (bf16*)carve(65536 * 2);
    bf16* Wbf_q  = (bf16*)carve(65536 * 2);
    bf16* Wbf_k  = (bf16*)carve(65536 * 2);
    bf16* Wbf_v  = (bf16*)carve(65536 * 2);
    bf16* Wbf_f1 = (bf16*)carve(262144 * 2);
    bf16* Wbf_f2 = (bf16*)carve(262144 * 2);

    auto gemm = [&](float* C32, bf16* C16, const bf16* A, const bf16* Bw,
                    const float* bias, int M, int Nd, int Kd, int relu) {
        dim3 g(Nd / 128, M / 128);
        gemm_bf16_wmma<<<g, 256, 0, stream>>>(C32, C16, A, Bw, bias, M, Nd, Kd, relu);
    };
    auto cvt = [&](bf16* dst, const float* src, long n) {
        const long n4 = n / 4;
        cvt_bf16_kernel<<<(unsigned)((n4 + 255) / 256), 256, 0, stream>>>(dst, src, n4);
    };

    // ---------------- input embedding (tiny K: VALU) ----------------
    rowlinear_kernel<<<(BNROWS * 64 + 255) / 256, 256, 0, stream>>>(
        E1, x, P[1], P[2], P[3], P[4], 1, BNROWS, 3, 64);
    rowlinear_kernel<<<(BNROWS * 128 + 255) / 256, 256, 0, stream>>>(
        E2, E1, P[5], P[6], P[7], P[8], 1, BNROWS, 64, 128);
    cvt(E2bf, E2, (long)BNROWS * 128);

    // ---------------- transformer blocks ----------------
    const bf16* hin_bf = E2bf;
    int inC = 128;
    for (int blk = 0; blk < 4; ++blk) {
        const float* const* Pb = &P[9 + blk * 18];

        // stage this block's weights as bf16
        cvt(Wbf_in, Pb[0],  (long)256 * inC);
        cvt(Wbf_q,  Pb[4],  65536);
        cvt(Wbf_k,  Pb[6],  65536);
        cvt(Wbf_v,  Pb[8],  65536);
        cvt(Wbf_f1, Pb[12], 262144);
        cvt(Wbf_f2, Pb[14], 262144);

        // h = LN(hin @ Win^T + bin)  (also mirror h to bf16)
        gemm(T, nullptr, hin_bf, Wbf_in, Pb[1], BNROWS, 256, inC, 0);
        layernorm_kernel<<<BNROWS / 8, 256, 0, stream>>>(H, Hbf, T, nullptr, Pb[2], Pb[3]);

        // KNN in feature space: per batch, dot = H_b @ H_b^T then top-16
        rowsumsq_kernel<<<BNROWS / 8, 256, 0, stream>>>(SQ, H);
        for (int b = 0; b < B_BATCH; ++b) {
            const bf16* Hb = Hbf + (size_t)b * NPTS * CDIM;
            gemm(BIG, nullptr, Hb, Hb, nullptr, NPTS, NPTS, 256, 0);
            topk_kernel<<<NPTS, 256, 0, stream>>>(
                IDX + (size_t)b * NPTS * KNN, BIG, SQ + (size_t)b * NPTS);
        }

        // Q, K, V linears
        gemm(Qb, nullptr, Hbf, Wbf_q, Pb[5], BNROWS, 256, 256, 0);
        gemm(Kb, nullptr, Hbf, Wbf_k, Pb[7], BNROWS, 256, 256, 0);
        gemm(Vb, nullptr, Hbf, Wbf_v, Pb[9], BNROWS, 256, 256, 0);

        // gathered 16-neighbor attention
        attn_kernel<<<BNROWS, 128, 0, stream>>>(AO, Qb, Kb, Vb, IDX);

        // h2 = LN(attn + h)  (mirror to bf16 for FFN input)
        layernorm_kernel<<<BNROWS / 8, 256, 0, stream>>>(H2, H2bf, AO, H, Pb[10], Pb[11]);

        // FFN: relu(h2 @ Wf1^T + bf1) -> bf16 hidden; then @ Wf2^T + bf2 -> fp32
        gemm(nullptr, FHbf, H2bf, Wbf_f1, Pb[13], BNROWS, 1024, 256, 1);
        gemm(T, nullptr, FHbf, Wbf_f2, Pb[15], BNROWS, 256, 1024, 0);

        // h = LN(ffn + h2)  (mirror to bf16 for next block)
        layernorm_kernel<<<BNROWS / 8, 256, 0, stream>>>(H, Hbf, T, H2, Pb[16], Pb[17]);

        hin_bf = Hbf;
        inC = 256;
    }

    // ---------------- global max-pool + FC head ----------------
    maxpool_kernel<<<(B_BATCH * CDIM + 255) / 256, 256, 0, stream>>>(GP, H);
    rowlinear_kernel<<<(B_BATCH * 512 + 255) / 256, 256, 0, stream>>>(
        F1, GP, P[81], P[82], P[83], P[84], 1, B_BATCH, 256, 512);
    rowlinear_kernel<<<(B_BATCH * 256 + 255) / 256, 256, 0, stream>>>(
        F2, F1, P[85], P[86], P[87], P[88], 1, B_BATCH, 512, 256);
    rowlinear_kernel<<<(B_BATCH * 40 + 255) / 256, 256, 0, stream>>>(
        (float*)d_out, F2, P[89], P[90], nullptr, nullptr, 0, B_BATCH, 256, 40);
}
